// MultiheadAttention_86835648790602
// MI455X (gfx1250) — matrix operand order repaired and checked by reading, not compiled, not run on hardware
//
#include <hip/hip_runtime.h>
#include <math.h>

// ---------------------------------------------------------------------------
// Problem constants (from reference)
// ---------------------------------------------------------------------------
constexpr int Tt = 1024, Bb = 32, Ee = 512, Hh = 8, HD = 64, KD = 16, Ss = 8;
constexpr int ROWS = Bb * Tt;            // 32768 token rows
constexpr int NCAT = 128 + 512 + 128;    // [Wk | Wv | Wq] = 768 fused output cols
constexpr int COL_K = 0, COL_V = 128, COL_Q = 640;

typedef __bf16 bf16;
typedef __attribute__((ext_vector_type(16))) __bf16 v16bf;
typedef __attribute__((ext_vector_type(8)))  __bf16 v8bf;
typedef __attribute__((ext_vector_type(8)))  float  v8f;

// ---------------------------------------------------------------------------
// Workspace layout (bytes)
// ---------------------------------------------------------------------------
constexpr size_t OFF_WINT   = 0;                                   // 512x512 bf16
constexpr size_t OFF_WCATT  = OFF_WINT  + (size_t)512*512*2;       // 768x512 bf16
constexpr size_t OFF_BIASC  = OFF_WCATT + (size_t)768*512*2;       // 768 f32
constexpr size_t OFF_XIN    = OFF_BIASC + (size_t)768*4 + 192;     // ROWSx512 bf16 (align)
constexpr size_t OFF_X      = OFF_XIN   + (size_t)ROWS*512*2;      // ROWSx512 bf16
constexpr size_t OFF_Y      = OFF_X     + (size_t)ROWS*512*2;      // ROWSx768 f32
constexpr size_t OFF_Q1     = OFF_Y     + (size_t)ROWS*768*4;      // 256x128 f32
constexpr size_t OFF_ATT    = OFF_Q1    + (size_t)256*128*4;       // 256x512 f32
constexpr size_t OFF_NMPRE  = OFF_ATT   + (size_t)256*512*4;       // 256x512 f32
constexpr size_t OFF_RACC   = OFF_NMPRE + (size_t)256*512*4;       // 32x512 f32
constexpr size_t OFF_NM2    = OFF_RACC  + (size_t)32*512*4;        // 256x512 f32
constexpr size_t OFF_K2     = OFF_NM2   + (size_t)256*512*4;       // 256x128 f32
constexpr size_t OFF_V2     = OFF_K2    + (size_t)256*128*4;       // 256x512 f32

// ---------------------------------------------------------------------------
// CDNA5 async global->LDS copy (ASYNCcnt path) + waits
// Generic pointers to __shared__ objects carry the LDS byte offset in their
// low 32 bits (flat-LDS aperture translation).
// ---------------------------------------------------------------------------
__device__ __forceinline__ void async_b128(unsigned lds_off, const void* g) {
  asm volatile("global_load_async_to_lds_b128 %0, %1, off"
               :: "v"(lds_off), "v"((unsigned long long)(uintptr_t)g)
               : "memory");
}
__device__ __forceinline__ void wait_async0() {
  asm volatile("s_wait_asynccnt 0x0" ::: "memory");
}
__device__ __forceinline__ void wait_async6() {
  asm volatile("s_wait_asynccnt 0x6" ::: "memory");
}
__device__ __forceinline__ unsigned lds_off32(const void* p) {
  return (unsigned)(uintptr_t)p;
}

// ---------------------------------------------------------------------------
// Weight convert / transpose:  Wt[n*K + k] = (bf16) W[k*N + n]
// ---------------------------------------------------------------------------
__global__ __launch_bounds__(256) void convT_kernel(const float* __restrict__ W,
                                                    bf16* __restrict__ Wt,
                                                    int K, int N) {
  int idx = blockIdx.x * 256 + threadIdx.x;
  if (idx >= K * N) return;
  int n = idx % N, k = idx / N;
  Wt[(size_t)n * K + k] = (bf16)W[(size_t)k * N + n];
}

__global__ __launch_bounds__(256) void biascat_kernel(const float* bk, const float* bv,
                                                      const float* bq, float* bc) {
  int n = blockIdx.x * 256 + threadIdx.x;
  if (n >= NCAT) return;
  bc[n] = (n < 128) ? bk[n] : (n < 640 ? bv[n - 128] : bq[n - 640]);
}

// key (T,B,E) f32  ->  Xin (B*T, E) bf16   (row r = b*T + t)
__global__ __launch_bounds__(256) void convXin_kernel(const float* __restrict__ key,
                                                      bf16* __restrict__ Xin) {
  size_t idx = (size_t)blockIdx.x * 256 + threadIdx.x;   // < ROWS*512
  int e = (int)(idx & 511);
  size_t r = idx >> 9;
  int b = (int)(r >> 10), t = (int)(r & 1023);
  Xin[idx] = (bf16)key[(((size_t)t * Bb) + b) * Ee + e];
}

// ---------------------------------------------------------------------------
// bf16 WMMA GEMM:  C[M,N] = A[M,K=512] * Wt[N,K]^T + bias
// block = 128 threads = 4 waves; block tile 128(M) x 64(N); wave tile 32x64
// (2 A fragments x 4 B fragments -> 8 WMMA per K-step, B reused 2x).
// K pipelined in chunks of 32 through TRIPLE-buffered LDS filled with
// GLOBAL_LOAD_ASYNC_TO_LDS_B128 (2 chunks in flight); fragment layouts per
// CDNA5 ISA 7.12.2.
// ---------------------------------------------------------------------------
constexpr int GK   = 512;     // K dimension (both GEMMs)
constexpr int KC   = 32;      // K chunk per pipeline stage
constexpr int NCH  = GK / KC; // 16 chunks
constexpr int RS   = KC + 8;  // LDS row stride in bf16 (80B -> 20-bank step)

__global__ __launch_bounds__(128) void gemm_bf16_wmma(const bf16* __restrict__ A,
                                                      const bf16* __restrict__ Wt,
                                                      const float* __restrict__ bias,
                                                      bf16* __restrict__ Cb,
                                                      float* __restrict__ Cf,
                                                      int N, int outF32) {
  __shared__ bf16 sA[3][128 * RS];   // 30720 B
  __shared__ bf16 sB[3][64 * RS];    // 15360 B

  const int tid  = threadIdx.x;
  const int lane = tid & 31;
  const int wave = tid >> 5;           // 0..3
  const int r16  = lane & 15;
  const int hi   = lane >> 4;          // lane group selects K sub-chunks
  const int mBase = blockIdx.x * 128;
  const int nBase = blockIdx.y * 64;

  // async-copy work assignment (per chunk: A 128x32, B 64x32 bf16)
  const int brow = tid >> 1;                 // 0..63, 2 threads/row, 32B each
  const int bcol = (tid & 1) * 16;           // element offset within chunk

  const bf16* gA = A  + (size_t)(mBase + tid) * GK;       // thread copies a row
  const bf16* gB = Wt + (size_t)(nBase + brow) * GK + bcol;
  const unsigned lA = lds_off32(&sA[0][0]) + (unsigned)(tid * RS) * 2u;
  const unsigned lB = lds_off32(&sB[0][0]) + (unsigned)(brow * RS + bcol) * 2u;
  const unsigned strideA = (unsigned)(128 * RS) * 2u;     // bytes between buffers
  const unsigned strideB = (unsigned)(64 * RS) * 2u;

  // issue chunk c into buffer c%3  (6 async instructions per wave per chunk)
  auto issue = [&](int c) {
    const int kk = c * KC;
    const unsigned buf = (unsigned)(c % 3);
    const unsigned ba = lA + buf * strideA;
    const unsigned bb = lB + buf * strideB;
#pragma unroll
    for (int i = 0; i < 4; ++i)          // A: 4 x b128 per thread (64B row chunk)
      async_b128(ba + i * 16, gA + kk + i * 8);
#pragma unroll
    for (int i = 0; i < 2; ++i)          // B: 2 x b128 per thread
      async_b128(bb + i * 16, gB + kk + i * 8);
  };

  v8f acc0[4] = {};
  v8f acc1[4] = {};

  issue(0);
  issue(1);
  for (int c = 0; c < NCH; ++c) {
    // Wait for chunk c to land: if a newer chunk is in flight, asynccnt<=6
    // guarantees the older one completed (in-order); on the last chunk wait 0.
    if (c + 1 < NCH) wait_async6(); else wait_async0();
    __syncthreads();                 // all waves: chunk c visible, buf (c+2)%3 free
    if (c + 2 < NCH) issue(c + 2);

    const int buf = c % 3;
    const bf16* pa = &sA[buf][(wave * 32 + r16) * RS];

    // A fragments 16x32: hi=0 -> K {0..7,16..23}; hi=1 -> K {8..15,24..31}
    v8bf a0l = *(const v8bf*)(pa + 8 * hi);
    v8bf a0h = *(const v8bf*)(pa + 16 + 8 * hi);
    v8bf a1l = *(const v8bf*)(pa + 16 * RS + 8 * hi);
    v8bf a1h = *(const v8bf*)(pa + 16 * RS + 16 + 8 * hi);
    v16bf af0, af1;
#pragma unroll
    for (int i = 0; i < 8; ++i) {
      af0[i] = a0l[i]; af0[8 + i] = a0h[i];
      af1[i] = a1l[i]; af1[8 + i] = a1h[i];
    }
#pragma unroll
    for (int j = 0; j < 4; ++j) {
      // B fragment 32x16: col = r16, hi=0 -> K {0..7,16..23}; hi=1 -> K {8..15,24..31}
      const bf16* pb = &sB[buf][(j * 16 + r16) * RS + 8 * hi];
      v8bf b_lo = *(const v8bf*)(pb);
      v8bf b_hi = *(const v8bf*)(pb + 16);
      v16bf bfv;
#pragma unroll
      for (int i = 0; i < 8; ++i) { bfv[i] = b_lo[i]; bfv[8 + i] = b_hi[i]; }
      acc0[j] = __builtin_amdgcn_wmma_f32_16x16x32_bf16(
          false, af0, false, bfv, (short)0, acc0[j], false, false);
      acc1[j] = __builtin_amdgcn_wmma_f32_16x16x32_bf16(
          false, af1, false, bfv, (short)0, acc1[j], false, false);
    }
  }

  // D layout: VGPR r -> row = base + r + 8*hi ; col = nBase + j*16 + r16
#pragma unroll
  for (int j = 0; j < 4; ++j) {
    int col = nBase + j * 16 + r16;
    float bc = bias[col];
#pragma unroll
    for (int r2 = 0; r2 < 8; ++r2) {
      int row0 = mBase + wave * 32 + r2 + 8 * hi;
      int row1 = row0 + 16;
      float v0 = acc0[j][r2] + bc;
      float v1 = acc1[j][r2] + bc;
      if (outF32) {
        Cf[(size_t)row0 * N + col] = v0;
        Cf[(size_t)row1 * N + col] = v1;
      } else {
        Cb[(size_t)row0 * N + col] = (bf16)v0;
        Cb[(size_t)row1 * N + col] = (bf16)v1;
      }
    }
  }
}

// ---------------------------------------------------------------------------
// q1 = memory @ Wq + bq   (256 rows x 128 cols, K=512)
// ---------------------------------------------------------------------------
__global__ __launch_bounds__(128) void q1_kernel(const float* __restrict__ memory,
                                                 const float* __restrict__ Wq,
                                                 const float* __restrict__ bq,
                                                 float* __restrict__ q1) {
  int row = blockIdx.x, tid = threadIdx.x;
  __shared__ float sm[512];
  for (int i = tid; i < 512; i += 128) sm[i] = memory[(size_t)row * 512 + i];
  __syncthreads();
  float acc = bq[tid];
  for (int e = 0; e < 512; ++e) acc += sm[e] * Wq[(size_t)e * 128 + tid];
  q1[(size_t)row * 128 + tid] = acc;
}

// ---------------------------------------------------------------------------
// Attention 1: memory slots query tokens; softmax over 1024 then top-3 mask.
// One block per (b,h). Y holds [k | v | q2] per token row.
// ---------------------------------------------------------------------------
__global__ __launch_bounds__(256) void attn1_kernel(const float* __restrict__ q1,
                                                    const float* __restrict__ Y,
                                                    float* __restrict__ attended) {
  const int bh = blockIdx.x, b = bh >> 3, h = bh & 7, tid = threadIdx.x;
  __shared__ float sq[128];
  __shared__ float slog[1024];
  __shared__ float sval[256];
  __shared__ int   sidx[256];
  __shared__ float top_v[3];
  __shared__ int   top_i[3];
  __shared__ float sdenom;

  if (tid < 128) sq[tid] = q1[(size_t)(b * 8 + (tid >> 4)) * 128 + h * 16 + (tid & 15)];
  __syncthreads();

  for (int s = 0; s < 8; ++s) {
    for (int t = tid; t < 1024; t += 256) {
      const float* kr = Y + (size_t)(b * Tt + t) * NCAT + COL_K + h * 16;
      float a = 0.f;
#pragma unroll
      for (int d = 0; d < 16; ++d) a += sq[s * 16 + d] * kr[d];
      slog[t] = a;
    }
    __syncthreads();

    // top-3 logits via 3 argmax passes
    for (int p = 0; p < 3; ++p) {
      float bv = -3.4e38f; int bi = -1;
      for (int t = tid; t < 1024; t += 256) {
        bool skip = false;
        for (int q = 0; q < p; ++q) skip |= (top_i[q] == t);
        float v = slog[t];
        if (!skip && v > bv) { bv = v; bi = t; }
      }
      sval[tid] = bv; sidx[tid] = bi;
      __syncthreads();
      for (int off = 128; off > 0; off >>= 1) {
        if (tid < off && sval[tid + off] > sval[tid]) {
          sval[tid] = sval[tid + off]; sidx[tid] = sidx[tid + off];
        }
        __syncthreads();
      }
      if (tid == 0) { top_v[p] = sval[0]; top_i[p] = sidx[0]; }
      __syncthreads();
    }

    // full softmax denominator (softmax over all 1024, then top-3 mask)
    float m1 = top_v[0];
    float ls = 0.f;
    for (int t = tid; t < 1024; t += 256) ls += __expf(slog[t] - m1);
    sval[tid] = ls;
    __syncthreads();
    for (int off = 128; off > 0; off >>= 1) {
      if (tid < off) sval[tid] += sval[tid + off];
      __syncthreads();
    }
    if (tid == 0) sdenom = sval[0];
    __syncthreads();

    if (tid < 64) {
      float inv = 1.f / sdenom;
      float p0 = __expf(top_v[0] - m1) * inv;
      float p1 = __expf(top_v[1] - m1) * inv;
      float p2 = __expf(top_v[2] - m1) * inv;
      const float* v0 = Y + (size_t)(b * Tt + top_i[0]) * NCAT + COL_V + h * 64;
      const float* v1 = Y + (size_t)(b * Tt + top_i[1]) * NCAT + COL_V + h * 64;
      const float* v2 = Y + (size_t)(b * Tt + top_i[2]) * NCAT + COL_V + h * 64;
      attended[(size_t)(b * 8 + s) * 512 + h * 64 + tid] =
          p0 * v0[tid] + p1 * v1[tid] + p2 * v2[tid];
    }
    __syncthreads();
  }
}

// ---------------------------------------------------------------------------
// Per-slot memory update: LN(mem+att) -> relu MLP -> LN -> nm_pre
// ---------------------------------------------------------------------------
__device__ inline float block_sum_512(float v, float* sred) {
  int tid = threadIdx.x;
  sred[tid] = v; __syncthreads();
  for (int off = 256; off > 0; off >>= 1) {
    if (tid < off) sred[tid] += sred[tid + off];
    __syncthreads();
  }
  float s = sred[0]; __syncthreads();
  return s;
}

__global__ __launch_bounds__(512) void memupdate_kernel(const float* __restrict__ memory,
                                                        const float* __restrict__ attended,
                                                        const float* __restrict__ W_mlp,
                                                        const float* __restrict__ b_mlp,
                                                        const float* __restrict__ g1,
                                                        const float* __restrict__ be1,
                                                        const float* __restrict__ g2,
                                                        const float* __restrict__ be2,
                                                        float* __restrict__ nm_pre) {
  int row = blockIdx.x, tid = threadIdx.x;
  __shared__ float sm[512];
  __shared__ float sred[512];
  float z = memory[(size_t)row * 512 + tid] + attended[(size_t)row * 512 + tid];
  float mu = block_sum_512(z, sred) * (1.f / 512.f);
  float var = block_sum_512((z - mu) * (z - mu), sred) * (1.f / 512.f);
  float mval = (z - mu) * rsqrtf(var + 1e-5f) * g1[tid] + be1[tid];
  sm[tid] = mval;
  __syncthreads();
  float acc = b_mlp[tid];
  for (int e = 0; e < 512; ++e) acc += sm[e] * W_mlp[(size_t)e * 512 + tid];
  float h2 = mval + fmaxf(acc, 0.f);
  float mu2 = block_sum_512(h2, sred) * (1.f / 512.f);
  float var2 = block_sum_512((h2 - mu2) * (h2 - mu2), sred) * (1.f / 512.f);
  nm_pre[(size_t)row * 512 + tid] = (h2 - mu2) * rsqrtf(var2 + 1e-5f) * g2[tid] + be2[tid];
}

// ---------------------------------------------------------------------------
// racc[b,e] = mean_t relu(w_rep[e] * X[b,t,e])
// ---------------------------------------------------------------------------
__global__ __launch_bounds__(256) void redmean_kernel(const bf16* __restrict__ X,
                                                      const float* __restrict__ w_rep,
                                                      float* __restrict__ racc) {
  int b = blockIdx.x;
  int e = blockIdx.y * 256 + threadIdx.x;
  float w = w_rep[e], acc = 0.f;
  const bf16* base = X + (size_t)b * Tt * 512 + e;
  for (int t = 0; t < Tt; ++t) acc += fmaxf(w * (float)base[(size_t)t * 512], 0.f);
  racc[b * 512 + e] = acc * (1.f / (float)Tt);
}

// ---------------------------------------------------------------------------
// Gates + final memory: one block per (b,s)
// ---------------------------------------------------------------------------
__global__ __launch_bounds__(256) void gates_kernel(const float* __restrict__ memory,
                                                    const float* __restrict__ racc,
                                                    const float* __restrict__ W_ig,
                                                    const float* __restrict__ b_ig,
                                                    const float* __restrict__ W_mg,
                                                    const float* __restrict__ b_mg,
                                                    const float* __restrict__ fb,
                                                    const float* __restrict__ ib,
                                                    const float* __restrict__ nm_pre,
                                                    float* __restrict__ nm2) {
  int row = blockIdx.x, b = row >> 3, s = row & 7, tid = threadIdx.x;
  __shared__ float stm[512];
  __shared__ float sr[512];
  for (int i = tid; i < 512; i += 256) {
    stm[i] = tanhf(memory[(size_t)row * 512 + i]);
    sr[i]  = racc[b * 512 + i];
  }
  __syncthreads();
  float forget_bias = fb[0], input_bias = ib[0];
  const float* wmg = W_mg + (size_t)s * 512 * 1024;
  for (int n = tid; n < 512; n += 256) {
    float ai = b_ig[n]       + b_mg[s * 1024 + n];
    float af = b_ig[512 + n] + b_mg[s * 1024 + 512 + n];
    for (int e = 0; e < 512; ++e) {
      float rv = sr[e], tv = stm[e];
      ai += rv * W_ig[(size_t)e * 1024 + n]       + tv * wmg[(size_t)e * 1024 + n];
      af += rv * W_ig[(size_t)e * 1024 + 512 + n] + tv * wmg[(size_t)e * 1024 + 512 + n];
    }
    float ig = 1.f / (1.f + __expf(-(ai + input_bias)));
    float fg = 1.f / (1.f + __expf(-(af + forget_bias)));
    nm2[(size_t)row * 512 + n] =
        ig * tanhf(nm_pre[(size_t)row * 512 + n]) + fg * memory[(size_t)row * 512 + n];
  }
}

// ---------------------------------------------------------------------------
// k2/v2 projections of new memory (256 rows, K=512)
// ---------------------------------------------------------------------------
__global__ __launch_bounds__(256) void kv2_kernel(const float* __restrict__ nm2,
                                                  const float* __restrict__ Wk,
                                                  const float* __restrict__ bk,
                                                  const float* __restrict__ Wv,
                                                  const float* __restrict__ bv,
                                                  float* __restrict__ k2,
                                                  float* __restrict__ v2) {
  int row = blockIdx.x, tid = threadIdx.x;
  __shared__ float sm[512];
  for (int i = tid; i < 512; i += 256) sm[i] = nm2[(size_t)row * 512 + i];
  __syncthreads();
  for (int j = tid; j < 640; j += 256) {
    if (j < 128) {
      float acc = bk[j];
      for (int e = 0; e < 512; ++e) acc += sm[e] * Wk[(size_t)e * 128 + j];
      k2[(size_t)row * 128 + j] = acc;
    } else {
      int jj = j - 128;
      float acc = bv[jj];
      for (int e = 0; e < 512; ++e) acc += sm[e] * Wv[(size_t)e * 512 + jj];
      v2[(size_t)row * 512 + jj] = acc;
    }
  }
}

// ---------------------------------------------------------------------------
// Attention 2: every token queries the 8 memory slots; write (T,B,E) output.
// One block per (b,t).
// ---------------------------------------------------------------------------
__global__ __launch_bounds__(256) void attn2_kernel(const float* __restrict__ Y,
                                                    const float* __restrict__ k2,
                                                    const float* __restrict__ v2,
                                                    float* __restrict__ out) {
  int row = blockIdx.x, b = row >> 10, t = row & 1023, tid = threadIdx.x;
  __shared__ float sq[128];
  __shared__ float sp[64];   // [h*8+s]
  if (tid < 128) sq[tid] = Y[(size_t)row * NCAT + COL_Q + tid];
  __syncthreads();
  if (tid < 64) {
    int h = tid >> 3, s = tid & 7;
    const float* kr = k2 + (size_t)(b * 8 + s) * 128 + h * 16;
    float a = 0.f;
#pragma unroll
    for (int d = 0; d < 16; ++d) a += sq[h * 16 + d] * kr[d];
    sp[tid] = a;
  }
  __syncthreads();
  if (tid < 8) {
    int h = tid;
    float m = -3.4e38f;
    for (int s = 0; s < 8; ++s) m = fmaxf(m, sp[h * 8 + s]);
    float den = 0.f;
    for (int s = 0; s < 8; ++s) { float e = __expf(sp[h * 8 + s] - m); sp[h * 8 + s] = e; den += e; }
    float inv = 1.f / den;
    for (int s = 0; s < 8; ++s) sp[h * 8 + s] *= inv;
  }
  __syncthreads();
  for (int e = tid; e < 512; e += 256) {
    int h = e >> 6, dd = e & 63;
    float acc = 0.f;
#pragma unroll
    for (int s = 0; s < 8; ++s)
      acc += sp[h * 8 + s] * v2[(size_t)(b * 8 + s) * 512 + h * 64 + dd];
    out[((size_t)t * Bb + b) * Ee + e] = acc;
  }
}

// ---------------------------------------------------------------------------
// Launch
// ---------------------------------------------------------------------------
extern "C" void kernel_launch(void* const* d_in, const int* in_sizes, int n_in,
                              void* d_out, int out_size, void* d_ws, size_t ws_size,
                              hipStream_t stream) {
  const float* key    = (const float*)d_in[1];
  const float* memory = (const float*)d_in[2];
  const float* W_in   = (const float*)d_in[3];
  const float* b_in   = (const float*)d_in[4];
  const float* Wq     = (const float*)d_in[5];
  const float* bq     = (const float*)d_in[6];
  const float* Wk     = (const float*)d_in[7];
  const float* bk     = (const float*)d_in[8];
  const float* Wv     = (const float*)d_in[9];
  const float* bv     = (const float*)d_in[10];
  const float* W_mlp  = (const float*)d_in[11];
  const float* b_mlp  = (const float*)d_in[12];
  const float* g1     = (const float*)d_in[13];
  const float* be1    = (const float*)d_in[14];
  const float* g2     = (const float*)d_in[15];
  const float* be2    = (const float*)d_in[16];
  const float* w_rep  = (const float*)d_in[17];
  const float* W_ig   = (const float*)d_in[18];
  const float* b_ig   = (const float*)d_in[19];
  const float* W_mg   = (const float*)d_in[20];
  const float* b_mg   = (const float*)d_in[21];
  const float* fb     = (const float*)d_in[22];
  const float* ib     = (const float*)d_in[23];

  char* ws = (char*)d_ws;
  bf16*  WinT   = (bf16*)(ws + OFF_WINT);
  bf16*  WcatT  = (bf16*)(ws + OFF_WCATT);
  float* biasc  = (float*)(ws + OFF_BIASC);
  bf16*  Xin    = (bf16*)(ws + OFF_XIN);
  bf16*  X      = (bf16*)(ws + OFF_X);
  float* Y      = (float*)(ws + OFF_Y);
  float* q1     = (float*)(ws + OFF_Q1);
  float* att    = (float*)(ws + OFF_ATT);
  float* nmpre  = (float*)(ws + OFF_NMPRE);
  float* racc   = (float*)(ws + OFF_RACC);
  float* nm2    = (float*)(ws + OFF_NM2);
  float* k2     = (float*)(ws + OFF_K2);
  float* v2     = (float*)(ws + OFF_V2);
  float* out    = (float*)d_out;

  // 1) weight prep
  convT_kernel<<<(512 * 512 + 255) / 256, 256, 0, stream>>>(W_in, WinT, 512, 512);
  convT_kernel<<<(512 * 128 + 255) / 256, 256, 0, stream>>>(Wk, WcatT + (size_t)COL_K * 512, 512, 128);
  convT_kernel<<<(512 * 512 + 255) / 256, 256, 0, stream>>>(Wv, WcatT + (size_t)COL_V * 512, 512, 512);
  convT_kernel<<<(512 * 128 + 255) / 256, 256, 0, stream>>>(Wq, WcatT + (size_t)COL_Q * 512, 512, 128);
  biascat_kernel<<<3, 256, 0, stream>>>(bk, bv, bq, biasc);

  // 2) key -> Xin (bf16, (B*T, E))
  convXin_kernel<<<(int)(((size_t)ROWS * 512) / 256), 256, 0, stream>>>(key, Xin);

  // 3) X = Xin @ W_in^T + b_in         (WMMA + async-LDS pipeline, bf16 out)
  gemm_bf16_wmma<<<dim3(ROWS / 128, 512 / 64), 128, 0, stream>>>(
      Xin, WinT, b_in, X, nullptr, 512, 0);

  // 4) Y = X @ [Wk|Wv|Wq]^T + biascat  (WMMA + async-LDS pipeline, f32 out)
  gemm_bf16_wmma<<<dim3(ROWS / 128, NCAT / 64), 128, 0, stream>>>(
      X, WcatT, biasc, nullptr, Y, NCAT, 1);

  // 5) q1 = memory @ Wq + bq
  q1_kernel<<<Bb * Ss, 128, 0, stream>>>(memory, Wq, bq, q1);

  // 6) attention over memory (top-3 write)
  attn1_kernel<<<Bb * Hh, 256, 0, stream>>>(q1, Y, att);

  // 7) LN -> MLP -> LN
  memupdate_kernel<<<Bb * Ss, 512, 0, stream>>>(memory, att, W_mlp, b_mlp,
                                                g1, be1, g2, be2, nmpre);

  // 8) gating
  redmean_kernel<<<dim3(Bb, 2), 256, 0, stream>>>(X, w_rep, racc);
  gates_kernel<<<Bb * Ss, 256, 0, stream>>>(memory, racc, W_ig, b_ig, W_mg, b_mg,
                                            fb, ib, nmpre, nm2);

  // 9) k2/v2 from updated memory
  kv2_kernel<<<Bb * Ss, 256, 0, stream>>>(nm2, Wk, bk, Wv, bv, k2, v2);

  // 10) broadcast read + transpose to (T,B,E)
  attn2_kernel<<<ROWS, 256, 0, stream>>>(Y, k2, v2, out);

  (void)in_sizes; (void)n_in; (void)out_size; (void)ws_size;
}
